// MultiHeadedAttention_8495445311959
// MI455X (gfx1250) — matrix operand order repaired and checked by reading, not compiled, not run on hardware
//
#include <hip/hip_runtime.h>

typedef __attribute__((ext_vector_type(16))) __bf16 v16bf;
typedef __attribute__((ext_vector_type(8)))  __bf16 v8bf;
typedef __attribute__((ext_vector_type(8)))  float  v8f;

#define B_SZ    4
#define T_SEQ   512
#define D_MODEL 768
#define NH      12
#define DKH     64
#define MROWS   (B_SZ * T_SEQ)        // 2048
#define SCALE_F 0.125f                // 1/sqrt(64)

// ---------- helpers ----------

__device__ __forceinline__ __bf16 f2bf(float f) {
    unsigned int u = __float_as_uint(f);
    u = u + 0x7FFFu + ((u >> 16) & 1u);           // round-to-nearest-even
    unsigned short s = (unsigned short)(u >> 16);
    __bf16 r;
    __builtin_memcpy(&r, &s, 2);
    return r;
}

// A-matrix 16x32 bf16 fragment (ISA 7.12.2): lane lo = row, kh = lane>>4;
// per-lane K = {kh*8 + 0..7} U {16 + kh*8 + 0..7}  -> two contiguous 16B loads
__device__ __forceinline__ v16bf load_a_frag(const __bf16* __restrict__ base,
                                             int ld, int row0, int k0,
                                             int lo, int kh) {
    const __bf16* p = base + (size_t)(row0 + lo) * ld + k0 + kh * 8;
    v8bf a = *(const v8bf*)p;
    v8bf b = *(const v8bf*)(p + 16);
    return __builtin_shufflevector(a, b, 0,1,2,3,4,5,6,7,8,9,10,11,12,13,14,15);
}

// B-matrix 32x16 bf16 fragment: lane lo = col n, kh = lane>>4;
// per-lane K = {kh*8 + 0..7} U {16 + kh*8 + 0..7} (Bt row-major [n][k])
__device__ __forceinline__ v16bf load_b_frag(const __bf16* __restrict__ base,
                                             int ld, int n0, int k0,
                                             int lo, int kh) {
    const __bf16* p = base + (size_t)(n0 + lo) * ld + k0 + kh * 8;
    v8bf a = *(const v8bf*)p;
    v8bf b = *(const v8bf*)(p + 16);
    return __builtin_shufflevector(a, b, 0,1,2,3,4,5,6,7,8,9,10,11,12,13,14,15);
}

__device__ __forceinline__ v8f wmma_bf16(v16bf a, v16bf b, v8f c) {
    return __builtin_amdgcn_wmma_f32_16x16x32_bf16(false, a, false, b,
                                                   (short)0, c, false, false);
}

// ---------- pre-pass kernels ----------

__global__ void cvt_f32_to_bf16(const float* __restrict__ src,
                                __bf16* __restrict__ dst, int n) {
    for (int i = blockIdx.x * blockDim.x + threadIdx.x; i < n;
         i += gridDim.x * blockDim.x)
        dst[i] = f2bf(src[i]);
}

// wt[n][k] = (bf16) w[k][n]   (dim x dim)
__global__ void transpose_to_bf16(const float* __restrict__ w,
                                  __bf16* __restrict__ wt, int dim) {
    int n = dim * dim;
    for (int i = blockIdx.x * blockDim.x + threadIdx.x; i < n;
         i += gridDim.x * blockDim.x) {
        int k = i / dim, c = i % dim;
        wt[(size_t)c * dim + k] = f2bf(w[i]);
    }
}

// temporal bias: bias[b][t][s] = sum_j leaky_relu(tm*w1[j]+b1[j])*w2[j] + b2
__global__ __launch_bounds__(256)
void tbias_kernel(const float* __restrict__ tmat,
                  const float* __restrict__ w1, const float* __restrict__ b1,
                  const float* __restrict__ w2, const float* __restrict__ b2,
                  float* __restrict__ out, int n) {
    __shared__ float sw1[64], sb1[64], sw2[64];
    if (threadIdx.x < 64) {
        sw1[threadIdx.x] = w1[threadIdx.x];
        sb1[threadIdx.x] = b1[threadIdx.x];
        sw2[threadIdx.x] = w2[threadIdx.x];
    }
    __syncthreads();
    float bb = b2[0];
    for (int i = blockIdx.x * blockDim.x + threadIdx.x; i < n;
         i += gridDim.x * blockDim.x) {
        float tm = 1.0f / logf(2.7182818284590452f + tmat[i]);
        float acc = bb;
        #pragma unroll
        for (int j = 0; j < 64; ++j) {
            float hv = fmaf(tm, sw1[j], sb1[j]);
            hv = hv > 0.0f ? hv : 0.2f * hv;
            acc = fmaf(hv, sw2[j], acc);
        }
        out[i] = acc;
    }
}

// ---------- GEMM: C[M][N] = A[M][K] * Bt[N][K]^T + bias ----------
// one wave computes a 16x64 tile. mode: 0 = fp32 row-major out,
// 1 = bf16 Q/K layout [bh][t][dk], 2 = bf16 V-transposed [bh][dk][t]
__global__ __launch_bounds__(32)
void gemm_bf16(const __bf16* __restrict__ A, const __bf16* __restrict__ Bt,
               const float* __restrict__ bias, void* __restrict__ out,
               int M, int N, int Kdim, int mode) {
    int ntiles = N / 64;
    int m0 = (blockIdx.x / ntiles) * 16;
    int n0 = (blockIdx.x % ntiles) * 64;
    int lane = threadIdx.x & 31, lo = lane & 15, kh = lane >> 4;

    v8f c0 = {}, c1 = {}, c2 = {}, c3 = {};
    for (int k0 = 0; k0 < Kdim; k0 += 32) {
        v16bf af  = load_a_frag(A,  Kdim, m0,      k0, lo, kh);
        v16bf bf0 = load_b_frag(Bt, Kdim, n0,      k0, lo, kh);
        v16bf bf1 = load_b_frag(Bt, Kdim, n0 + 16, k0, lo, kh);
        v16bf bf2 = load_b_frag(Bt, Kdim, n0 + 32, k0, lo, kh);
        v16bf bf3 = load_b_frag(Bt, Kdim, n0 + 48, k0, lo, kh);
        c0 = wmma_bf16(af, bf0, c0);
        c1 = wmma_bf16(af, bf1, c1);
        c2 = wmma_bf16(af, bf2, c2);
        c3 = wmma_bf16(af, bf3, c3);
    }

    #pragma unroll
    for (int j = 0; j < 8; ++j) {
        int m = m0 + j + 8 * kh;
        float vals[4] = { c0[j], c1[j], c2[j], c3[j] };
        #pragma unroll
        for (int q = 0; q < 4; ++q) {
            int n = n0 + q * 16 + lo;
            float v = vals[q] + bias[n];
            if (mode == 0) {
                ((float*)out)[(size_t)m * N + n] = v;
            } else {
                int bb = m / T_SEQ, t = m % T_SEQ;
                int hh = n / DKH,   dk = n % DKH;
                if (mode == 1)
                    ((__bf16*)out)[(((size_t)bb * NH + hh) * T_SEQ + t) * DKH + dk] = f2bf(v);
                else
                    ((__bf16*)out)[(((size_t)bb * NH + hh) * DKH + dk) * T_SEQ + t] = f2bf(v);
            }
        }
    }
}

// ---------- fused flash attention ----------
// one wave per (b, h, 16-row query tile)
__global__ __launch_bounds__(32)
void attn_kernel(const __bf16* __restrict__ Qbf,   // [bh][t][dk]
                 const __bf16* __restrict__ Kbf,   // [bh][s][dk]
                 const __bf16* __restrict__ Vt,    // [bh][dk][s]
                 const float* __restrict__ tbias,  // [b][t][s]
                 const unsigned char* __restrict__ pmask, // [b][t][s]
                 __bf16* __restrict__ attn_out) {  // [b*t][D]
    int tiles_per_bh = T_SEQ / 16;
    int bh = blockIdx.x / tiles_per_bh;
    int t0 = (blockIdx.x % tiles_per_bh) * 16;
    int b = bh / NH, h = bh % NH;

    int lane = threadIdx.x & 31, lo = lane & 15, kh = lane >> 4;

    __shared__ __attribute__((aligned(16))) __bf16 lds_p[16][32];

    const __bf16* Q = Qbf + (size_t)bh * T_SEQ * DKH;
    const __bf16* K = Kbf + (size_t)bh * T_SEQ * DKH;
    const __bf16* V = Vt  + (size_t)bh * DKH * T_SEQ;

    v16bf qf0 = load_a_frag(Q, DKH, t0, 0,  lo, kh);
    v16bf qf1 = load_a_frag(Q, DKH, t0, 32, lo, kh);

    v8f o0 = {}, o1 = {}, o2 = {}, o3 = {};
    float mrow[8], lrow[8];
    #pragma unroll
    for (int j = 0; j < 8; ++j) { mrow[j] = -1e30f; lrow[j] = 0.0f; }

    const float* brow = tbias + ((size_t)b * T_SEQ + t0) * T_SEQ;
    const unsigned char* prow = pmask + ((size_t)b * T_SEQ + t0) * T_SEQ;

    int s_end = (t0 == 0) ? T_SEQ : (t0 + 16);   // row 0 attends everywhere
    for (int sb = 0; sb < s_end; sb += 32) {
        // ---- scores: S(16x32) = Q(16x64) K^T ----
        v8f s0 = {}, s1 = {};
        v16bf kf;
        kf = load_b_frag(K, DKH, sb,      0,  lo, kh); s0 = wmma_bf16(qf0, kf, s0);
        kf = load_b_frag(K, DKH, sb,      32, lo, kh); s0 = wmma_bf16(qf1, kf, s0);
        kf = load_b_frag(K, DKH, sb + 16, 0,  lo, kh); s1 = wmma_bf16(qf0, kf, s1);
        kf = load_b_frag(K, DKH, sb + 16, 32, lo, kh); s1 = wmma_bf16(qf1, kf, s1);

        // ---- scale + temporal bias + padding + causal masks ----
        #pragma unroll
        for (int j = 0; j < 8; ++j) {
            int m = j + 8 * kh;
            int t = t0 + m;
            size_t off0 = (size_t)m * T_SEQ + sb + lo;
            size_t off1 = off0 + 16;
            float v0 = fmaf(s0[j], SCALE_F, brow[off0]);
            float v1 = fmaf(s1[j], SCALE_F, brow[off1]);
            if (!prow[off0]) v0 = -1e30f;
            if (!prow[off1]) v1 = -1e30f;
            if (t != 0) {
                if (sb + lo      > t) v0 = -1e30f;
                if (sb + 16 + lo > t) v1 = -1e30f;
            }
            s0[j] = v0; s1[j] = v1;
        }

        // ---- online softmax (rows live across 16 lanes of a half) ----
        #pragma unroll
        for (int j = 0; j < 8; ++j) {
            float mx = fmaxf(s0[j], s1[j]);
            #pragma unroll
            for (int msk = 1; msk < 16; msk <<= 1)
                mx = fmaxf(mx, __shfl_xor(mx, msk, 32));
            float mnew = fmaxf(mrow[j], mx);
            float corr = __expf(mrow[j] - mnew);
            mrow[j] = mnew;
            float p0 = __expf(s0[j] - mnew);
            float p1 = __expf(s1[j] - mnew);
            s0[j] = p0; s1[j] = p1;
            float rs = p0 + p1;
            #pragma unroll
            for (int msk = 1; msk < 16; msk <<= 1)
                rs += __shfl_xor(rs, msk, 32);
            lrow[j] = lrow[j] * corr + rs;
            o0[j] *= corr; o1[j] *= corr; o2[j] *= corr; o3[j] *= corr;
        }

        // ---- reshape P(16x32) through LDS into A-fragment layout ----
        #pragma unroll
        for (int j = 0; j < 8; ++j) {
            int m = j + 8 * kh;
            lds_p[m][lo]      = f2bf(s0[j]);
            lds_p[m][16 + lo] = f2bf(s1[j]);
        }
        __builtin_amdgcn_wave_barrier();
        v16bf pf = load_a_frag(&lds_p[0][0], 32, 0, 0, lo, kh);
        __builtin_amdgcn_wave_barrier();

        // ---- O += P(16x32) V(32x64): 4 dk-tiles ----
        v16bf vf;
        vf = load_b_frag(V, T_SEQ, 0,  sb, lo, kh); o0 = wmma_bf16(pf, vf, o0);
        vf = load_b_frag(V, T_SEQ, 16, sb, lo, kh); o1 = wmma_bf16(pf, vf, o1);
        vf = load_b_frag(V, T_SEQ, 32, sb, lo, kh); o2 = wmma_bf16(pf, vf, o2);
        vf = load_b_frag(V, T_SEQ, 48, sb, lo, kh); o3 = wmma_bf16(pf, vf, o3);
    }

    // ---- epilogue: divide by row sums, write (b*t, h*64+dk) bf16 ----
    #pragma unroll
    for (int j = 0; j < 8; ++j) {
        int m = j + 8 * kh;
        int t = t0 + m;
        float inv = 1.0f / lrow[j];
        size_t row = ((size_t)b * T_SEQ + t) * D_MODEL + h * DKH;
        attn_out[row + 0  + lo] = f2bf(o0[j] * inv);
        attn_out[row + 16 + lo] = f2bf(o1[j] * inv);
        attn_out[row + 32 + lo] = f2bf(o2[j] * inv);
        attn_out[row + 48 + lo] = f2bf(o3[j] * inv);
    }
}

// ---------- host launch ----------

extern "C" void kernel_launch(void* const* d_in, const int* in_sizes, int n_in,
                              void* d_out, int out_size, void* d_ws, size_t ws_size,
                              hipStream_t stream) {
    const float* x      = (const float*)d_in[0];
    const unsigned char* pmask = (const unsigned char*)d_in[1];
    const float* tmat   = (const float*)d_in[2];
    const float* wq     = (const float*)d_in[3];
    const float* bq     = (const float*)d_in[4];
    const float* wk     = (const float*)d_in[5];
    const float* bk     = (const float*)d_in[6];
    const float* wv     = (const float*)d_in[7];
    const float* bv     = (const float*)d_in[8];
    const float* wproj  = (const float*)d_in[9];
    const float* bproj  = (const float*)d_in[10];
    const float* tb1_w  = (const float*)d_in[11];
    const float* tb1_b  = (const float*)d_in[12];
    const float* tb2_w  = (const float*)d_in[13];
    const float* tb2_b  = (const float*)d_in[14];

    // workspace layout (all chunks multiple of 256B)
    constexpr size_t XBF   = (size_t)MROWS * D_MODEL * 2;          // 3 MB
    constexpr size_t WT    = (size_t)D_MODEL * D_MODEL * 2;        // 1.125 MB
    constexpr size_t QK    = (size_t)B_SZ * NH * T_SEQ * DKH * 2;  // 3 MB
    constexpr size_t TB    = (size_t)B_SZ * T_SEQ * T_SEQ * 4;     // 4 MB

    char* w = (char*)d_ws;
    __bf16* xbf  = (__bf16*)w;               w += XBF;
    __bf16* wqT  = (__bf16*)w;               w += WT;
    __bf16* wkT  = (__bf16*)w;               w += WT;
    __bf16* wvT  = (__bf16*)w;               w += WT;
    __bf16* wpT  = (__bf16*)w;               w += WT;
    __bf16* Qbf  = (__bf16*)w;               w += QK;
    __bf16* Kbf  = (__bf16*)w;               w += QK;
    __bf16* Vtb  = (__bf16*)w;               w += QK;
    float*  tbia = (float*)w;                w += TB;
    __bf16* attn = (__bf16*)w;               w += XBF;

    // pre-pass conversions
    int nx = MROWS * D_MODEL;
    cvt_f32_to_bf16<<<(nx + 255) / 256, 256, 0, stream>>>(x, xbf, nx);
    int nw = D_MODEL * D_MODEL;
    int gw = (nw + 255) / 256;
    transpose_to_bf16<<<gw, 256, 0, stream>>>(wq,    wqT, D_MODEL);
    transpose_to_bf16<<<gw, 256, 0, stream>>>(wk,    wkT, D_MODEL);
    transpose_to_bf16<<<gw, 256, 0, stream>>>(wv,    wvT, D_MODEL);
    transpose_to_bf16<<<gw, 256, 0, stream>>>(wproj, wpT, D_MODEL);

    // temporal bias (head-independent -> compute once)
    tbias_kernel<<<1024, 256, 0, stream>>>(tmat, tb1_w, tb1_b, tb2_w, tb2_b,
                                           tbia, B_SZ * T_SEQ * T_SEQ);

    // QKV projections (one wave per 16x64 tile)
    int ggrid = (MROWS / 16) * (D_MODEL / 64);  // 1536
    gemm_bf16<<<ggrid, 32, 0, stream>>>(xbf, wqT, bq, Qbf, MROWS, D_MODEL, D_MODEL, 1);
    gemm_bf16<<<ggrid, 32, 0, stream>>>(xbf, wkT, bk, Kbf, MROWS, D_MODEL, D_MODEL, 1);
    gemm_bf16<<<ggrid, 32, 0, stream>>>(xbf, wvT, bv, Vtb, MROWS, D_MODEL, D_MODEL, 2);

    // fused flash attention: wave per (b,h,16-row tile)
    attn_kernel<<<B_SZ * NH * (T_SEQ / 16), 32, 0, stream>>>(
        Qbf, Kbf, Vtb, tbia, pmask, attn);

    // output projection -> fp32 d_out
    gemm_bf16<<<ggrid, 32, 0, stream>>>(attn, wpT, bproj, d_out,
                                        MROWS, D_MODEL, D_MODEL, 0);
}
